// LegendreMemoryUnit_2534030705263
// MI455X (gfx1250) — matrix operand order repaired and checked by reading, not compiled, not run on hardware
//
#include <hip/hip_runtime.h>

typedef __attribute__((ext_vector_type(16))) _Float16 v16h;
typedef __attribute__((ext_vector_type(8)))  _Float16 v8h;
typedef __attribute__((ext_vector_type(8)))  float    v8f;

static constexpr int kB  = 8;
static constexpr int kS  = 2048;
static constexpr int kDI = 512;
static constexpr int kDM = 1024;
static constexpr int kROWS = kB * kS;          // 16384 (B*S flattened GEMM rows)
static constexpr int kNC   = 64;               // chunks per sequence
static constexpr int kLCH  = 32;               // chunk length (kNC*kLCH == kS)
static constexpr int kCH   = kB * kDM;         // 8192 independent recurrence channels

#define DECAY 0.9f
#define INPF  0.1f
#define DECAY_L 0.03433683820292512f           /* 0.9^32 */

// ---------------------------------------------------------------------------
// CDNA5 async global->LDS copy (ASYNCcnt path), inline asm (portable across
// toolchains; the clang builtin has toolchain-specific signatures).
// VDST = LDS byte address VGPR, VADDR = 64-bit global address pair, GV mode.
// ---------------------------------------------------------------------------
__device__ __forceinline__ void async_copy_b128(const _Float16* g, _Float16* l) {
    unsigned loff = (unsigned)(unsigned long long)l;   // flat LDS addr[31:0] = LDS offset
    asm volatile("global_load_async_to_lds_b128 %0, %1, off"
                 :: "v"(loff), "v"(g) : "memory");
}

__device__ __forceinline__ void wait_async0() {
    asm volatile("s_wait_asynccnt 0x0" ::: "memory");
}

// ---------------------------------------------------------------------------
// f32 -> f16 conversion (x, Wi, Wo)
// ---------------------------------------------------------------------------
__global__ void cvt_f32_f16(const float* __restrict__ in, _Float16* __restrict__ out, size_t n) {
    size_t i = (size_t)blockIdx.x * blockDim.x + threadIdx.x;
    if (i < n) out[i] = (_Float16)in[i];
}

// ---------------------------------------------------------------------------
// Tiled WMMA GEMM:  C[Mtot x Ntot] = A[Mtot x K] * W[Ntot x K]^T + bias[Ntot]
// A, W in f16; accumulate f32; OUT_T = f16 (GEMM1 -> u) or f32 (GEMM2 -> out).
// Block tile 128x128, BK=32, 256 threads = 8 waves, wave tile 64x32 (4x2 frags).
// Double-buffered LDS filled by GLOBAL_LOAD_ASYNC_TO_LDS_B128.
// ---------------------------------------------------------------------------
template <typename OUT_T>
__global__ __launch_bounds__(256)
void gemm_wmma_f16(const _Float16* __restrict__ A,
                   const _Float16* __restrict__ W,
                   const float*    __restrict__ bias,
                   OUT_T*          __restrict__ C,
                   int K, int Ntot)
{
    __shared__ _Float16 sA[2][128 * 40];   // stride 40 halves (80B) -> conflict-free b128
    __shared__ _Float16 sB[2][128 * 40];   // W tile, N-major / K-contiguous

    const int tid    = threadIdx.x;
    const int lane   = tid & 31;
    const int lane16 = lane & 15;
    const int laneHi = (lane >> 4) & 1;
    const int wave   = tid >> 5;
    const int waveM  = (wave >> 2) * 64;    // 0 or 64
    const int waveN  = (wave & 3)  * 32;    // 0,32,64,96
    const long blockM = (long)blockIdx.x * 128;
    const long blockN = (long)blockIdx.y * 128;

    // This thread's two tile chunks: l in {tid, tid+256}; row = l/4, col = (l%4)*8
    const int l0r = tid >> 2,         l0c = (tid & 3) << 3;
    const int l1r = (tid + 256) >> 2, l1c = ((tid + 256) & 3) << 3;

    auto issue_tile = [&](int buf, int k0) {
        async_copy_b128(&A[(blockM + l0r) * (long)K + k0 + l0c], &sA[buf][l0r * 40 + l0c]);
        async_copy_b128(&W[(blockN + l0r) * (long)K + k0 + l0c], &sB[buf][l0r * 40 + l0c]);
        async_copy_b128(&A[(blockM + l1r) * (long)K + k0 + l1c], &sA[buf][l1r * 40 + l1c]);
        async_copy_b128(&W[(blockN + l1r) * (long)K + k0 + l1c], &sB[buf][l1r * 40 + l1c]);
    };

    v8f acc[4][2] = {};

    const int nk = K >> 5;
    issue_tile(0, 0);
    wait_async0();
    __syncthreads();

    for (int kt = 0; kt < nk; ++kt) {
        const int buf = kt & 1;
        if (kt + 1 < nk) issue_tile(buf ^ 1, (kt + 1) << 5);

        // A fragment (ISA 16-bit A 16x32 layout):
        // lanes 0-15 (M=lane): v0..3 = K 0..7,  v4..7 = K 16..23
        // lanes16-31 (M=lane-16): v0..3 = K 8..15, v4..7 = K 24..31
        v16h af[4];
        #pragma unroll
        for (int fm = 0; fm < 4; ++fm) {
            const _Float16* p = &sA[buf][(waveM + fm * 16 + lane16) * 40 + laneHi * 8];
            v8h lo = *(const v8h*)(p);
            v8h hi = *(const v8h*)(p + 16);
            af[fm] = __builtin_shufflevector(lo, hi, 0,1,2,3,4,5,6,7,8,9,10,11,12,13,14,15);
        }

        // B fragment (16-bit B 32x16): lane16 = N column; lanes 0-15 hold K 0..7, 16..23,
        // lanes 16-31 hold K 8..15, 24..31 (same k order as A).
        v16h bf[2];
        #pragma unroll
        for (int fn = 0; fn < 2; ++fn) {
            const _Float16* p = &sB[buf][(waveN + fn * 16 + lane16) * 40 + laneHi * 8];
            v8h lo = *(const v8h*)(p);
            v8h hi = *(const v8h*)(p + 16);
            bf[fn] = __builtin_shufflevector(lo, hi, 0,1,2,3,4,5,6,7,8,9,10,11,12,13,14,15);
        }

        #pragma unroll
        for (int fm = 0; fm < 4; ++fm)
            #pragma unroll
            for (int fn = 0; fn < 2; ++fn)
                acc[fm][fn] = __builtin_amdgcn_wmma_f32_16x16x32_f16(
                    false, af[fm], false, bf[fn], (short)0, acc[fm][fn], false, false);

        // Next tile fully in LDS (own async ops) + all waves done reading this one.
        if (kt + 1 < nk) wait_async0();
        __syncthreads();
    }

    // Epilogue. C frag layout: VGPR v -> row (laneHi?8:0)+v, col = lane16.
    #pragma unroll
    for (int fn = 0; fn < 2; ++fn) {
        const long col = blockN + waveN + fn * 16 + lane16;
        const float bv = bias[col];
        #pragma unroll
        for (int fm = 0; fm < 4; ++fm) {
            #pragma unroll
            for (int v = 0; v < 8; ++v) {
                const long row = blockM + waveM + fm * 16 + laneHi * 8 + v;
                C[row * (long)Ntot + col] = (OUT_T)(acc[fm][fn][v] + bv);
            }
        }
    }
}

// ---------------------------------------------------------------------------
// Recurrence phase 1: per-chunk local end state with zero init.
// g = j*kCH + c ; c = b*kDM + m
// ---------------------------------------------------------------------------
__global__ void chunk_sum_kernel(const _Float16* __restrict__ u, float* __restrict__ e) {
    int g = blockIdx.x * blockDim.x + threadIdx.x;
    int c = g & (kCH - 1);
    int j = g >> 13;                 // / kCH
    int b = c >> 10;                 // / kDM
    int m = c & (kDM - 1);
    size_t idx = ((size_t)b * kS + (size_t)j * kLCH) * kDM + m;
    float h = 0.0f;
    #pragma unroll 4
    for (int k = 0; k < kLCH; ++k) {
        h = DECAY * h + INPF * (float)u[idx];
        idx += kDM;
    }
    e[g] = h;
}

// ---------------------------------------------------------------------------
// Recurrence phase 2: sequential carry propagation across chunks (per channel);
// emits chunk-start states and final_memory.
// ---------------------------------------------------------------------------
__global__ void chunk_scan_kernel(const float* __restrict__ e,
                                  const float* __restrict__ mem0,
                                  float* __restrict__ Sst,
                                  float* __restrict__ finalMem) {
    int c = blockIdx.x * blockDim.x + threadIdx.x;   // [0, kCH)
    float s = mem0[c];
    #pragma unroll 8
    for (int j = 0; j < kNC; ++j) {
        Sst[(size_t)j * kCH + c] = s;
        s = DECAY_L * s + e[(size_t)j * kCH + c];
    }
    finalMem[c] = s;
}

// ---------------------------------------------------------------------------
// Recurrence phase 3: re-scan each chunk seeded with its carry; write mems(f16).
// ---------------------------------------------------------------------------
__global__ void chunk_apply_kernel(const _Float16* __restrict__ u,
                                   const float* __restrict__ Sst,
                                   _Float16* __restrict__ mems) {
    int g = blockIdx.x * blockDim.x + threadIdx.x;
    int c = g & (kCH - 1);
    int j = g >> 13;
    int b = c >> 10;
    int m = c & (kDM - 1);
    size_t idx = ((size_t)b * kS + (size_t)j * kLCH) * kDM + m;
    float h = Sst[g];
    #pragma unroll 4
    for (int k = 0; k < kLCH; ++k) {
        h = DECAY * h + INPF * (float)u[idx];
        mems[idx] = (_Float16)h;
        idx += kDM;
    }
}

// ---------------------------------------------------------------------------
extern "C" void kernel_launch(void* const* d_in, const int* in_sizes, int n_in,
                              void* d_out, int out_size, void* d_ws, size_t ws_size,
                              hipStream_t stream) {
    const float* x    = (const float*)d_in[0];   // [B,S,DI]
    const float* mem0 = (const float*)d_in[1];   // [B,DM]
    const float* Wi   = (const float*)d_in[2];   // [DM,DI]
    const float* bi   = (const float*)d_in[3];   // [DM]
    const float* Wo   = (const float*)d_in[4];   // [DI,DM]
    const float* bo   = (const float*)d_in[5];   // [DI]

    float* out      = (float*)d_out;                       // [B,S,DI]
    float* finalMem = out + (size_t)kROWS * kDI;           // [B,DM]

    // Workspace carve-out (~90 MB total)
    char* ws = (char*)d_ws;
    size_t off = 0;
    auto alloc = [&](size_t bytes) -> char* {
        char* p = ws + off;
        off += (bytes + 255) & ~(size_t)255;
        return p;
    };
    _Float16* x16  = (_Float16*)alloc((size_t)kROWS * kDI * 2);   // 16.8 MB
    _Float16* Wi16 = (_Float16*)alloc((size_t)kDM * kDI * 2);     //  1 MB
    _Float16* Wo16 = (_Float16*)alloc((size_t)kDI * kDM * 2);     //  1 MB
    _Float16* u16  = (_Float16*)alloc((size_t)kROWS * kDM * 2);   // 33.5 MB
    _Float16* m16  = (_Float16*)alloc((size_t)kROWS * kDM * 2);   // 33.5 MB
    float*    e    = (float*)alloc((size_t)kCH * kNC * 4);        //  2 MB
    float*    Sst  = (float*)alloc((size_t)kCH * kNC * 4);        //  2 MB

    // Precision downcast for WMMA inputs
    {
        size_t nx = (size_t)kROWS * kDI;
        cvt_f32_f16<<<dim3((unsigned)((nx + 255) / 256)), 256, 0, stream>>>(x, x16, nx);
        size_t nw = (size_t)kDM * kDI;
        cvt_f32_f16<<<dim3((unsigned)((nw + 255) / 256)), 256, 0, stream>>>(Wi, Wi16, nw);
        cvt_f32_f16<<<dim3((unsigned)((nw + 255) / 256)), 256, 0, stream>>>(Wo, Wo16, nw);
    }

    // GEMM1: u = x * Wi^T + bi   (16384 x 1024, K=512) -> f16
    gemm_wmma_f16<_Float16><<<dim3(kROWS / 128, kDM / 128), 256, 0, stream>>>(
        x16, Wi16, bi, u16, kDI, kDM);

    // Chunked linear recurrence: mem_t = 0.9 mem_{t-1} + 0.1 u_t
    chunk_sum_kernel<<<dim3(kCH * kNC / 256), 256, 0, stream>>>(u16, e);
    chunk_scan_kernel<<<dim3(kCH / 256), 256, 0, stream>>>(e, mem0, Sst, finalMem);
    chunk_apply_kernel<<<dim3(kCH * kNC / 256), 256, 0, stream>>>(u16, Sst, m16);

    // GEMM2: out = mems * Wo^T + bo  (16384 x 512, K=1024) -> f32
    gemm_wmma_f16<float><<<dim3(kROWS / 128, kDI / 128), 256, 0, stream>>>(
        m16, Wo16, bo, out, kDM, kDI);
}
